// Reservoir_57380763074897
// MI455X (gfx1250) — matrix operand order repaired and checked by reading, not compiled, not run on hardware
//
#include <hip/hip_runtime.h>

#define FEATURES 128
#define NEURONS  512
#define THETA    256
#define TDIM     2048
#define K1       896           // K order: x(0..511) | s(512..767) | I(768..895)
#define KT1      28            // K1 / 32
#define KT2      8             // THETA / 32
#define NT1      16            // THETA / 16
#define NT2      32            // NEURONS / 16
#define AH_STRIDE 904          // 896 + 8 halves pad (16B aligned rows, spreads banks)

typedef _Float16 v16h __attribute__((ext_vector_type(16)));
typedef _Float16 v8h  __attribute__((ext_vector_type(8)));
typedef float    v8f  __attribute__((ext_vector_type(8)));

// Global (addrspace 1) pointer types so laundered pointers still lower to
// global_load_b128 (LOADcnt only) instead of flat_load_b128 (LOADcnt+DScnt).
typedef const __attribute__((address_space(1))) _Float16* gch;
typedef const __attribute__((address_space(1))) v16h*     gcv16h;

union AFrag { v16h v; v8h h[2]; };

// Pre-swizzle combined W1 = [W_ih_x | W_hh | W_ih_I] (256 x 896) into fp16
// B-fragment order: offset = ((nt*KT1 + kt)*32 + lane)*16 + e,
// where k = kt*32 + (e&7) + (e>=8 ? 16 : 0) + (lane>=16 ? 8 : 0), n = nt*16 + (lane&15).
__global__ void prep_w1(const float* __restrict__ W_ih,
                        const float* __restrict__ W_hh,
                        _Float16* __restrict__ W1h) {
    int o = blockIdx.x * 256 + threadIdx.x;
    if (o >= NT1 * KT1 * 32 * 16) return;
    int e    = o & 15;
    int lane = (o >> 4) & 31;
    int tile = o >> 9;                 // nt*KT1 + kt
    int kt   = tile % KT1;
    int nt   = tile / KT1;
    int k = kt * 32 + (e & 7) + ((e >> 3) << 4) + ((lane >> 4) << 3);
    int n = nt * 16 + (lane & 15);
    float v;
    if (k < NEURONS)                 v = W_ih[n * (FEATURES + NEURONS) + k];
    else if (k < NEURONS + THETA)    v = W_hh[n * THETA + (k - NEURONS)];
    else                             v = W_ih[n * (FEATURES + NEURONS) + NEURONS + (k - (NEURONS + THETA))];
    W1h[o] = (_Float16)v;
}

// Pre-swizzle W_out (512 x 256) the same way (B = W_out^T, K=256, N=512).
__global__ void prep_w2(const float* __restrict__ W_out,
                        _Float16* __restrict__ W2h) {
    int o = blockIdx.x * 256 + threadIdx.x;
    if (o >= NT2 * KT2 * 32 * 16) return;
    int e    = o & 15;
    int lane = (o >> 4) & 31;
    int tile = o >> 9;                 // nt*KT2 + kt
    int kt   = tile & (KT2 - 1);
    int nt   = tile >> 3;
    int k = kt * 32 + (e & 7) + ((e >> 3) << 4) + ((lane >> 4) << 3);
    int n = nt * 16 + (lane & 15);
    W2h[o] = (_Float16)W_out[n * THETA + k];
}

// Persistent recurrent kernel: one block = 16 batch rows, 16 waves, 2048 steps.
__global__ __launch_bounds__(512)
void reservoir_persist(const float* __restrict__ Iin,
                       const float* __restrict__ b_ih,
                       const float* __restrict__ b_hh,
                       const float* __restrict__ b_out,
                       const float* __restrict__ Avec,
                       const _Float16* __restrict__ W1h,
                       const _Float16* __restrict__ W2h,
                       float* __restrict__ out) {
    __shared__ __align__(16) _Float16 Ah[16 * AH_STRIDE]; // fp16 activations [x|s|I]
    __shared__ float bias1[THETA];
    __shared__ float bo[NEURONS];
    __shared__ float Aa[NEURONS];

    const int tid    = threadIdx.x;
    const int wave   = tid >> 5;
    const int lane   = tid & 31;
    const int lh     = lane & 15;      // N column within tile / A-fragment row
    const int hi     = lane >> 4;      // half-wave select
    const int wgBase = blockIdx.x * 16;

    for (int q = tid; q < 16 * AH_STRIDE; q += 512) Ah[q] = (_Float16)0.0f;
    for (int q = tid; q < THETA;   q += 512) bias1[q] = b_ih[q] + b_hh[q];
    for (int q = tid; q < NEURONS; q += 512) { bo[q] = b_out[q]; Aa[q] = Avec[q]; }
    __syncthreads();
    for (int q = tid; q < 16 * FEATURES; q += 512) {     // I_0
        int m = q >> 7, c = q & 127;
        Ah[m * AH_STRIDE + (NEURONS + THETA) + c] =
            (_Float16)Iin[((wgBase + m) * TDIM + 0) * FEATURES + c];
    }
    __syncthreads();

    const int aRowOff = lh * AH_STRIDE;    // A fragment: M = lane&15
    const int aKoff   = hi * 8;            // 16-bit A layout: hi lanes hold K+8 / K+24

    // fp32 x state lives in registers: lane owns (m = r + 8*hi, n = (wave+16*tt)*16+lh)
    float xreg[2][8];
    #pragma unroll
    for (int tt = 0; tt < 2; ++tt)
        #pragma unroll
        for (int r = 0; r < 8; ++r) xreg[tt][r] = 0.0f;

    // Launderable weight bases (prevents LICM from hoisting 44 fragments out of
    // the 2048-step loop and spilling them to scratch).
    const _Float16* w1base = W1h;
    const _Float16* w2base = W2h;

    for (int t = 0; t < TDIM; ++t) {
        asm volatile("" : "+s"(w1base), "+s"(w2base));
        // Re-assert global address space after laundering -> global_load_b128.
        gch w1g = (gch)w1base;
        gch w2g = (gch)w2base;

        // ---- GEMM1: s_pre tile nt = wave, K = 896, software-pipelined ----
        gch w1p = w1g + ((wave * KT1) * 32 + lane) * 16;
        const _Float16* ap = Ah + aRowOff + aKoff;
        v8f acc1 = {};
        AFrag a0; v16h b0;
        a0.h[0] = *(const v8h*)(ap);
        a0.h[1] = *(const v8h*)(ap + 16);
        b0 = *(gcv16h)(w1p);
        #pragma unroll
        for (int kt = 0; kt < KT1; ++kt) {
            AFrag an = a0; v16h bn = b0;
            if (kt + 1 < KT1) {
                an.h[0] = *(const v8h*)(ap + (kt + 1) * 32);
                an.h[1] = *(const v8h*)(ap + (kt + 1) * 32 + 16);
                bn = *(gcv16h)(w1p + (kt + 1) * 512);
            }
            acc1 = __builtin_amdgcn_wmma_f32_16x16x32_f16(false, a0.v, false, b0,
                                                          (short)0, acc1, false, false);
            a0 = an; b0 = bn;
        }
        __syncthreads();  // all waves done reading Ah (x/s/I of step t)

        // ---- bias + tanh, write s_new (fp16) into Ah cols [512,768) ----
        {
            int n = wave * 16 + lh;
            float bsv = bias1[n];
            #pragma unroll
            for (int r = 0; r < 8; ++r) {
                int m = r + hi * 8;
                float v = tanhf(acc1[r] + bsv);
                Ah[m * AH_STRIDE + NEURONS + n] = (_Float16)v;
            }
        }
        __syncthreads();  // s_new visible to every wave

        // ---- GEMM2: f_pre tiles nt2 = wave, wave+16; K = 256, pipelined ----
        gch w2a = w2g + ((wave * KT2) * 32 + lane) * 16;
        gch w2b = w2g + (((wave + 16) * KT2) * 32 + lane) * 16;
        const _Float16* ap2 = Ah + aRowOff + NEURONS + aKoff;
        v8f acc2a = {}, acc2b = {};
        AFrag c0; v16h ba0, bb0;
        c0.h[0] = *(const v8h*)(ap2);
        c0.h[1] = *(const v8h*)(ap2 + 16);
        ba0 = *(gcv16h)(w2a);
        bb0 = *(gcv16h)(w2b);
        #pragma unroll
        for (int kt = 0; kt < KT2; ++kt) {
            AFrag cn = c0; v16h ban = ba0, bbn = bb0;
            if (kt + 1 < KT2) {
                cn.h[0] = *(const v8h*)(ap2 + (kt + 1) * 32);
                cn.h[1] = *(const v8h*)(ap2 + (kt + 1) * 32 + 16);
                ban = *(gcv16h)(w2a + (kt + 1) * 512);
                bbn = *(gcv16h)(w2b + (kt + 1) * 512);
            }
            acc2a = __builtin_amdgcn_wmma_f32_16x16x32_f16(false, c0.v, false, ba0,
                                                           (short)0, acc2a, false, false);
            acc2b = __builtin_amdgcn_wmma_f32_16x16x32_f16(false, c0.v, false, bb0,
                                                           (short)0, acc2b, false, false);
            c0 = cn; ba0 = ban; bb0 = bbn;
        }

        // ---- relu + implicit-Euler update; x state fp32 in regs ----
        #pragma unroll
        for (int tt = 0; tt < 2; ++tt) {
            v8f acc = tt ? acc2b : acc2a;
            int n = (wave + 16 * tt) * 16 + lh;
            float bov = bo[n], av = Aa[n];
            #pragma unroll
            for (int r = 0; r < 8; ++r) {
                int m  = r + hi * 8;
                float f  = fmaxf(acc[r] + bov, 0.0f);
                float xn = (xreg[tt][r] + 0.1f * f * av) / (1.1f + 0.1f * f);
                xreg[tt][r] = xn;
                Ah[m * AH_STRIDE + n] = (_Float16)xn;
                out[((wgBase + m) * TDIM + t) * NEURONS + n] = xn;
            }
        }

        // ---- stage I_{t+1} (region disjoint from GEMM2 reads) ----
        if (t + 1 < TDIM) {
            for (int q = tid; q < 16 * FEATURES; q += 512) {
                int m = q >> 7, c = q & 127;
                Ah[m * AH_STRIDE + (NEURONS + THETA) + c] =
                    (_Float16)Iin[((wgBase + m) * TDIM + (t + 1)) * FEATURES + c];
            }
        }
        // prefetch I_{t+2} into cache (global_prefetch_b8)
        if (t + 2 < TDIM && tid < 32) {
            int m = tid >> 1, halfline = tid & 1;
            __builtin_prefetch(&Iin[((wgBase + m) * TDIM + (t + 2)) * FEATURES + halfline * 64], 0, 3);
        }
        __syncthreads();  // x_next / I_{t+1} visible before next GEMM1
    }
}

extern "C" void kernel_launch(void* const* d_in, const int* in_sizes, int n_in,
                              void* d_out, int out_size, void* d_ws, size_t ws_size,
                              hipStream_t stream) {
    const float* I     = (const float*)d_in[0];
    const float* W_ih  = (const float*)d_in[1];
    const float* W_hh  = (const float*)d_in[2];
    const float* b_ih  = (const float*)d_in[3];
    const float* b_hh  = (const float*)d_in[4];
    const float* W_out = (const float*)d_in[5];
    const float* b_out = (const float*)d_in[6];
    const float* A     = (const float*)d_in[7];
    float* out = (float*)d_out;
    (void)in_sizes; (void)n_in; (void)out_size; (void)ws_size;

    _Float16* W1h = (_Float16*)d_ws;                                    // 448 KB
    _Float16* W2h = (_Float16*)((char*)d_ws +
                     (size_t)(NT1 * KT1 * 32 * 16) * sizeof(_Float16)); // +256 KB

    prep_w1<<<(NT1 * KT1 * 32 * 16 + 255) / 256, 256, 0, stream>>>(W_ih, W_hh, W1h);
    prep_w2<<<(NT2 * KT2 * 32 * 16 + 255) / 256, 256, 0, stream>>>(W_out, W2h);
    reservoir_persist<<<4, 512, 0, stream>>>(I, b_ih, b_hh, b_out, A, W1h, W2h, out);
}
